// BaseAttention_34909494182250
// MI455X (gfx1250) — matrix operand order repaired and checked by reading, not compiled, not run on hardware
//
#include <hip/hip_runtime.h>

typedef unsigned short u16;
typedef __attribute__((ext_vector_type(16))) __bf16 bf16x16;
typedef __attribute__((ext_vector_type(8)))  float  floatx8;
typedef __attribute__((ext_vector_type(4)))  int    i32x4;

union Frag { uint4 q[2]; bf16x16 v; };

__device__ __forceinline__ floatx8 wmma_bf16(bf16x16 a, bf16x16 b, floatx8 c) {
  return __builtin_amdgcn_wmma_f32_16x16x32_bf16(false, a, false, b, (short)0, c,
                                                 false, false);
}

__device__ __forceinline__ u16 f2bf(float f) {
  union { float f; unsigned u; } cvt; cvt.f = f;
  unsigned u = cvt.u;
  u += 0x7FFFu + ((u >> 16) & 1u);   // round-to-nearest-even
  return (u16)(u >> 16);
}

// ---------------------------------------------------------------------------
// CDNA5 async global->LDS copy (ASYNCcnt path), with synchronous fallback
// ---------------------------------------------------------------------------
#if __has_builtin(__builtin_amdgcn_global_load_async_to_lds_b128)
#define HAS_ASYNC_LDS 1
#else
#define HAS_ASYNC_LDS 0
#endif

__device__ __forceinline__ void async_copy16(u16* lds, const u16* g) {
#if HAS_ASYNC_LDS
  __builtin_amdgcn_global_load_async_to_lds_b128(
      (__attribute__((address_space(1))) i32x4*)g,
      (__attribute__((address_space(3))) i32x4*)lds, 0, 0);
#else
  *(uint4*)lds = *(const uint4*)g;
#endif
}

__device__ __forceinline__ void wait_async() {
#if HAS_ASYNC_LDS
#if __has_builtin(__builtin_amdgcn_s_wait_asynccnt)
  __builtin_amdgcn_s_wait_asynccnt(0);
#else
  asm volatile("s_wait_asynccnt 0" ::: "memory");
#endif
#endif
}

// ---------------------------------------------------------------------------
// fp32 -> bf16 (vectorized x4)
// ---------------------------------------------------------------------------
__global__ __launch_bounds__(256) void cvt_f32_bf16(const float* __restrict__ in,
                                                    u16* __restrict__ out, int n4) {
  int i = blockIdx.x * blockDim.x + threadIdx.x;
  if (i < n4) {
    float4 v = ((const float4*)in)[i];
    union { u16 u[4]; uint2 d; } o;
    o.u[0] = f2bf(v.x); o.u[1] = f2bf(v.y); o.u[2] = f2bf(v.z); o.u[3] = f2bf(v.w);
    ((uint2*)out)[i] = o.d;
  }
}

// ---------------------------------------------------------------------------
// bf16 GEMM:  Y[M,Nout] = alpha * X[M,K] @ W[Nout,K]^T
// block: 256 threads (8 waves), tile 128x128, K-chunk 32, double-buffered LDS
// wave grid 4(M) x 2(N); wave tile 32x64 = 2x4 16x16 frags
// ---------------------------------------------------------------------------
#define GP 40   // LDS row pitch in u16 (80B) for a 32-element K-chunk row

__global__ __launch_bounds__(256) void gemm_bf16(const u16* __restrict__ X,
                                                 const u16* __restrict__ W,
                                                 void* __restrict__ Y,
                                                 int M, int Nout, int K,
                                                 float alpha, int out_bf16) {
  __shared__ u16 Xs[2][128 * GP];
  __shared__ u16 Ws[2][128 * GP];

  const int tid  = threadIdx.x;
  const int lane = tid & 31;
  const int wave = tid >> 5;
  const int wm   = wave & 3;           // 0..3  -> 32-row slot
  const int wn   = wave >> 2;          // 0..1  -> 64-col slot
  const int m0   = blockIdx.x * 128;
  const int n0   = blockIdx.y * 128;

  const int lrow  = lane & 15;
  const int kbA   = (lane < 16) ? 0 : 8;    // A-frag K base (interleaved layout)
  const int kbB   = (lane < 16) ? 0 : 8;    // B-frag K base (interleaved layout)
  const int rbase = (lane >> 4) * 8;        // C-frag row base

  floatx8 acc[2][4];
#pragma unroll
  for (int a = 0; a < 2; ++a)
#pragma unroll
    for (int b = 0; b < 4; ++b) acc[a][b] = {};

  const int ldrow  = tid >> 1;   // 0..127
  const int ldhalf = tid & 1;    // 16 bf16 per half

  auto load_tile = [&](int k0, int buf) {
    const u16* xs = X + (size_t)(m0 + ldrow) * K + k0 + ldhalf * 16;
    async_copy16(&Xs[buf][ldrow * GP + ldhalf * 16], xs);
    async_copy16(&Xs[buf][ldrow * GP + ldhalf * 16 + 8], xs + 8);
    const u16* wsrc = W + (size_t)(n0 + ldrow) * K + k0 + ldhalf * 16;
    async_copy16(&Ws[buf][ldrow * GP + ldhalf * 16], wsrc);
    async_copy16(&Ws[buf][ldrow * GP + ldhalf * 16 + 8], wsrc + 8);
  };

  load_tile(0, 0);
  int buf = 0;
  for (int k0 = 0; k0 < K; k0 += 32, buf ^= 1) {
    wait_async();
    __syncthreads();
    if (k0 + 32 < K) load_tile(k0 + 32, buf ^ 1);

    Frag a[2];
#pragma unroll
    for (int mf = 0; mf < 2; ++mf) {
      const u16* p = &Xs[buf][(wm * 32 + mf * 16 + lrow) * GP + kbA];
      a[mf].q[0] = *(const uint4*)p;
      a[mf].q[1] = *(const uint4*)(p + 16);
    }
#pragma unroll
    for (int nf = 0; nf < 4; ++nf) {
      const u16* p = &Ws[buf][(wn * 64 + nf * 16 + lrow) * GP + kbB];
      Frag bf;
      bf.q[0] = *(const uint4*)p;
      bf.q[1] = *(const uint4*)(p + 16);
#pragma unroll
      for (int mf = 0; mf < 2; ++mf)
        acc[mf][nf] = wmma_bf16(a[mf].v, bf.v, acc[mf][nf]);
    }
  }

#pragma unroll
  for (int mf = 0; mf < 2; ++mf)
#pragma unroll
    for (int nf = 0; nf < 4; ++nf) {
      int col = n0 + wn * 64 + nf * 16 + lrow;
#pragma unroll
      for (int r = 0; r < 8; ++r) {
        int row = m0 + wm * 32 + mf * 16 + rbase + r;
        float v = acc[mf][nf][r] * alpha;
        if (out_bf16) ((u16*)Y)[(size_t)row * Nout + col] = f2bf(v);
        else          ((float*)Y)[(size_t)row * Nout + col] = v;
      }
    }
}

// ---------------------------------------------------------------------------
// Transpose V (cols 1024..2047 of KV buffer) -> Vt[B*1024][2048]
// ---------------------------------------------------------------------------
__global__ __launch_bounds__(256) void transpose_v(const u16* __restrict__ KV,
                                                   u16* __restrict__ Vt) {
  __shared__ u16 tile[32][33];
  const int c0 = blockIdx.x * 32, n0 = blockIdx.y * 32, b = blockIdx.z;
  const int tx = threadIdx.x, ty = threadIdx.y;  // 32 x 8
#pragma unroll
  for (int i = 0; i < 4; ++i) {
    int n = n0 + ty + i * 8;
    tile[ty + i * 8][tx] = KV[(size_t)(b * 2048 + n) * 2048 + 1024 + c0 + tx];
  }
  __syncthreads();
#pragma unroll
  for (int i = 0; i < 4; ++i) {
    int c = c0 + ty + i * 8;
    Vt[(size_t)(b * 1024 + c) * 2048 + n0 + tx] = tile[tx][ty + i * 8];
  }
}

// ---------------------------------------------------------------------------
// Flash attention (causal, online softmax), bf16 WMMA
// grid: (N/64, B*H), block: 128 threads (4 waves, 16 query rows each)
// K/V tiles double-buffered via async global->LDS copies
// ---------------------------------------------------------------------------
#define FN 2048
#define FD 1024
#define KP 72   // K-tile LDS pitch (u16) for 64-wide rows
#define VP 40   // Vt/P LDS pitch (u16) for 32-wide rows

__global__ __launch_bounds__(128) void flash_attn(const u16* __restrict__ Q,
                                                  const u16* __restrict__ KV,
                                                  const u16* __restrict__ Vt,
                                                  u16* __restrict__ Aout) {
  __shared__ u16 Kls[2][32 * KP];     // [32 keys][64 dh]
  __shared__ u16 Vls[2][64 * VP];     // [64 dh][32 keys]
  __shared__ u16 Pls[4 * 16 * VP];    // per-wave [16 rows][32 keys]

  const int tid  = threadIdx.x;
  const int lane = tid & 31;
  const int wave = tid >> 5;
  const int b    = blockIdx.y >> 4;
  const int h    = blockIdx.y & 15;
  const int q0   = blockIdx.x * 64;

  const int lrow  = lane & 15;
  const int kbA   = (lane < 16) ? 0 : 8;
  const int kbB   = (lane < 16) ? 0 : 8;
  const int rbase = (lane >> 4) * 8;

  // Q fragments for this wave's 16 rows (already scaled by dh^-0.5 in GEMM)
  Frag qf[2];
  {
    const u16* qp = Q + (size_t)(b * FN + q0 + wave * 16 + lrow) * FD + h * 64 + kbA;
    qf[0].q[0] = *(const uint4*)qp;
    qf[0].q[1] = *(const uint4*)(qp + 16);
    qf[1].q[0] = *(const uint4*)(qp + 32);
    qf[1].q[1] = *(const uint4*)(qp + 48);
  }

  floatx8 o[4];
#pragma unroll
  for (int f = 0; f < 4; ++f) o[f] = {};
  float mrow[8], lsum[8];
#pragma unroll
  for (int r = 0; r < 8; ++r) { mrow[r] = -1e30f; lsum[r] = 0.f; }

  auto load_kv = [&](int j, int buf) {
#pragma unroll
    for (int p = 0; p < 2; ++p) {
      int i = p * 128 + tid;
      int kr = i >> 3, ks = i & 7;
      const u16* src = KV + (size_t)(b * FN + j * 32 + kr) * 2048 + h * 64 + ks * 8;
      async_copy16(&Kls[buf][kr * KP + ks * 8], src);
      int vr = i >> 2, vs = i & 3;
      const u16* vsrc = Vt + (size_t)(b * 1024 + h * 64 + vr) * FN + j * 32 + vs * 8;
      async_copy16(&Vls[buf][vr * VP + vs * 8], vsrc);
    }
  };

  const int ntiles = (q0 + 64) / 32;
  load_kv(0, 0);
  int buf = 0;
  for (int j = 0; j < ntiles; ++j, buf ^= 1) {
    wait_async();
    __syncthreads();
    if (j + 1 < ntiles) load_kv(j + 1, buf ^ 1);

    // S = Q @ K^T  -> two 16x16 f32 frags (keys 0..15, 16..31 of tile)
    floatx8 s0 = {}, s1 = {};
#pragma unroll
    for (int c = 0; c < 2; ++c) {
      const u16* kp0 = &Kls[buf][lrow * KP + c * 32 + kbB];
      Frag b0; b0.q[0] = *(const uint4*)kp0; b0.q[1] = *(const uint4*)(kp0 + 16);
      const u16* kp1 = &Kls[buf][(lrow + 16) * KP + c * 32 + kbB];
      Frag b1; b1.q[0] = *(const uint4*)kp1; b1.q[1] = *(const uint4*)(kp1 + 16);
      s0 = wmma_bf16(qf[c].v, b0.v, s0);
      s1 = wmma_bf16(qf[c].v, b1.v, s1);
    }

    // causal mask + online softmax (row = rbase + r, key = j*32 + n)
    const int key0 = j * 32 + lrow;
    const int key1 = key0 + 16;
    float sc[8];
#pragma unroll
    for (int r = 0; r < 8; ++r) {
      int qrow = q0 + wave * 16 + rbase + r;
      float v0 = (key0 > qrow) ? -1e30f : s0[r];
      float v1 = (key1 > qrow) ? -1e30f : s1[r];
      float t = fmaxf(v0, v1);
      t = fmaxf(t, __shfl_xor(t, 1));
      t = fmaxf(t, __shfl_xor(t, 2));
      t = fmaxf(t, __shfl_xor(t, 4));
      t = fmaxf(t, __shfl_xor(t, 8));
      float mn = fmaxf(mrow[r], t);
      float scale = __expf(mrow[r] - mn);
      float p0 = __expf(v0 - mn);
      float p1 = __expf(v1 - mn);
      float ps = p0 + p1;
      ps += __shfl_xor(ps, 1);
      ps += __shfl_xor(ps, 2);
      ps += __shfl_xor(ps, 4);
      ps += __shfl_xor(ps, 8);
      lsum[r] = lsum[r] * scale + ps;
      mrow[r] = mn;
      sc[r]   = scale;
      s0[r] = p0;
      s1[r] = p1;
    }
#pragma unroll
    for (int f = 0; f < 4; ++f)
#pragma unroll
      for (int r = 0; r < 8; ++r) o[f][r] *= sc[r];

    // P (C-frag layout) -> LDS -> A-frag layout
    u16* pl = Pls + wave * 16 * VP;
#pragma unroll
    for (int r = 0; r < 8; ++r) {
      pl[(rbase + r) * VP + lrow]      = f2bf(s0[r]);
      pl[(rbase + r) * VP + lrow + 16] = f2bf(s1[r]);
    }
    asm volatile("s_wait_dscnt 0" ::: "memory");
    Frag pf;
    {
      const u16* pp = pl + lrow * VP + kbA;
      pf.q[0] = *(const uint4*)pp;
      pf.q[1] = *(const uint4*)(pp + 16);
    }

    // O += P @ V   (4 dh-frags of 16)
#pragma unroll
    for (int f = 0; f < 4; ++f) {
      const u16* vp = &Vls[buf][(f * 16 + lrow) * VP + kbB];
      Frag bv; bv.q[0] = *(const uint4*)vp; bv.q[1] = *(const uint4*)(vp + 16);
      o[f] = wmma_bf16(pf.v, bv.v, o[f]);
    }
  }

  // epilogue: out = O / (lsum + eps), bf16
#pragma unroll
  for (int f = 0; f < 4; ++f)
#pragma unroll
    for (int r = 0; r < 8; ++r) {
      float den = lsum[r] + 1e-8f;
      float val = o[f][r] / den;
      int row = b * FN + q0 + wave * 16 + rbase + r;
      Aout[(size_t)row * FD + h * 64 + f * 16 + lrow] = f2bf(val);
    }
}

// ---------------------------------------------------------------------------
// Host launcher
// ---------------------------------------------------------------------------
extern "C" void kernel_launch(void* const* d_in, const int* in_sizes, int n_in,
                              void* d_out, int out_size, void* d_ws, size_t ws_size,
                              hipStream_t stream) {
  const float* x   = (const float*)d_in[0];  // [2,2048,1024]
  const float* Wq  = (const float*)d_in[1];  // [1024,1024]
  const float* Wkv = (const float*)d_in[2];  // [2048,1024]
  const float* Wo  = (const float*)d_in[3];  // [1024,1024]
  float* out = (float*)d_out;                // [2,2048,1024]

  u16* ws = (u16*)d_ws;
  u16* xbf   = ws;                 // 4096x1024
  u16* wqbf  = xbf   + 4194304;    // 1024x1024
  u16* wkvbf = wqbf  + 1048576;    // 2048x1024
  u16* wobf  = wkvbf + 2097152;    // 1024x1024
  u16* qbuf  = wobf  + 1048576;    // 4096x1024
  u16* kvbuf = qbuf  + 4194304;    // 4096x2048
  u16* vt    = kvbuf + 8388608;    // 2048x2048
  u16* attn  = vt    + 4194304;    // 4096x1024

  cvt_f32_bf16<<<4096, 256, 0, stream>>>(x,   xbf,   1048576);
  cvt_f32_bf16<<<1024, 256, 0, stream>>>(Wq,  wqbf,   262144);
  cvt_f32_bf16<<<2048, 256, 0, stream>>>(Wkv, wkvbf,  524288);
  cvt_f32_bf16<<<1024, 256, 0, stream>>>(Wo,  wobf,   262144);

  // Q = x @ Wq^T * dh^-0.5   (dh = 64 -> 0.125)
  gemm_bf16<<<dim3(32, 8),  256, 0, stream>>>(xbf, wqbf,  qbuf,  4096, 1024, 1024,
                                              0.125f, 1);
  // KV = x @ Wkv^T
  gemm_bf16<<<dim3(32, 16), 256, 0, stream>>>(xbf, wkvbf, kvbuf, 4096, 2048, 1024,
                                              1.0f, 1);
  // Vt[b*1024+c][n] = V[b][n][c]
  transpose_v<<<dim3(32, 64, 2), dim3(32, 8), 0, stream>>>(kvbuf, vt);
  // attention
  flash_attn<<<dim3(32, 32), 128, 0, stream>>>(qbuf, kvbuf, vt, attn);
  // out = attn @ Wo^T (fp32 output)
  gemm_bf16<<<dim3(32, 8),  256, 0, stream>>>(attn, wobf, out, 4096, 1024, 1024,
                                              1.0f, 0);
}
